// AdaptivePoolingClassifier_43422119363005
// MI455X (gfx1250) — matrix operand order repaired and checked by reading, compiled, not run on hardware
//
#include <hip/hip_runtime.h>
#include <hip/hip_bf16.h>

typedef __attribute__((ext_vector_type(16))) _Float16 v16h;
typedef __attribute__((ext_vector_type(8)))  _Float16 v8h;
typedef __attribute__((ext_vector_type(8)))  float    v8f;
typedef __attribute__((ext_vector_type(4))) unsigned int u32x4;
typedef __attribute__((ext_vector_type(8))) int          i32x8;
typedef __attribute__((ext_vector_type(4))) int          i32x4;

union Pack16 { v16h v; v8h h8[2]; };

#define FEAT 128
#define WAVES_PER_BLOCK 4
#define ROWS_PER_WAVE 32   // two 16-row WMMA tiles per wave

// d_ws byte offsets
#define WS_ACCUM 0        // 16 floats: S[0..7], T[8..15]
#define WS_WT1   256      // [128][384] f16
#define WS_WT2   98560    // [384][256] f16
#define WS_WT3   295168   // [256][128] f16
#define WS_WT4   360704   // [128][16]  f16 (padded from 5 cols)

// ---------------- prep kernels ----------------

__global__ void init_accum_kernel(float* accum) {
    if (threadIdx.x < 16) accum[threadIdx.x] = 0.0f;
}

// W: [G][K] f32 (row-major)  ->  Wt: [K][Gpad] f16, Wt[k][g] = W[g][k], zero-pad g>=G
__global__ void convert_wt_kernel(const float* __restrict__ W, _Float16* __restrict__ Wt,
                                  int G, int K, int Gpad) {
    int idx = blockIdx.x * 256 + threadIdx.x;
    if (idx >= K * Gpad) return;
    int k = idx / Gpad;
    int g = idx - k * Gpad;
    Wt[idx] = (g < G) ? (_Float16)W[(size_t)g * K + k] : (_Float16)0.0f;
}

// ---------------- TDM: DMA a [rows][cols] f32 tile from global to LDS ----------------

__device__ __forceinline__ void tdm_load_tile_f32(const float* gsrc, float* ldst,
                                                  unsigned rows, unsigned cols) {
#if __has_builtin(__builtin_amdgcn_tensor_load_to_lds)
    const uint64_t ga = (uint64_t)(uintptr_t)gsrc;
    const unsigned la = (unsigned)(uintptr_t)ldst;    // generic addr low 32 = LDS byte offset

    u32x4 g0;
    g0[0] = 1u;                                              // count=1, user mode, no gather
    g0[1] = la;                                              // lds_addr
    g0[2] = (unsigned)ga;                                    // global_addr[31:0]
    g0[3] = (unsigned)((ga >> 32) & 0x01FFFFFFu) | (2u << 30); // global_addr[56:32], type=2

    i32x8 g1;
    g1[0] = (int)(2u << 16);                                 // data_size=2 (4 bytes)
    g1[1] = (int)((cols & 0xFFFFu) << 16);                   // tensor_dim0[15:0]  @ bits 63:48
    g1[2] = (int)(((cols >> 16) & 0xFFFFu) |                 // tensor_dim0[31:16] @ bits 79:64
                  ((rows & 0xFFFFu) << 16));                 // tensor_dim1[15:0]  @ bits 95:80
    g1[3] = (int)(((rows >> 16) & 0xFFFFu) |                 // tensor_dim1[31:16]
                  ((cols & 0xFFFFu) << 16));                 // tile_dim0 @ bits 127:112
    g1[4] = (int)(rows & 0xFFFFu);                           // tile_dim1 (tile_dim2 = 0)
    g1[5] = (int)cols;                                       // tensor_dim0_stride[31:0]
    g1[6] = 0;                                               // stride0[47:32], stride1[15:0]
    g1[7] = 0;

    i32x4 z4 = {0, 0, 0, 0};
    i32x8 z8 = {0, 0, 0, 0, 0, 0, 0, 0};
    __asm__ volatile("" ::: "memory");
    __builtin_amdgcn_tensor_load_to_lds(g0, g1, z4, z4, z8, 0);  // 6-arg toolchain form
#if __has_builtin(__builtin_amdgcn_s_wait_tensorcnt)
    __builtin_amdgcn_s_wait_tensorcnt(0);
#else
    __asm__ volatile("s_wait_tensorcnt 0x0" ::: "memory");
#endif
    __asm__ volatile("" ::: "memory");
#else
    // Fallback: per-lane vector copy (raw f32).
    const int lane = threadIdx.x & 31;
    const float4* xp = (const float4*)(gsrc + (size_t)lane * cols);
    float4* dp = (float4*)ldst + lane * (cols / 4);
    for (unsigned i = 0; i < cols / 4; ++i) dp[i] = xp[i];
#endif
}

// ---------------- A-fragment loaders (ISA 16-bit A layout) ----------------

template<int K>
__device__ __forceinline__ v16h load_afrag(const _Float16* base, int m, int hh, int kb) {
    const _Float16* ap = base + m * K + kb * 32 + hh * 8;
    Pack16 pa;
    pa.h8[0] = *(const v8h*)ap;          // K = kb*32 + hh*8 + [0..7]
    pa.h8[1] = *(const v8h*)(ap + 16);   // K = kb*32 + 16 + hh*8 + [0..7]
    return pa.v;
}

template<int K>
__device__ __forceinline__ v16h load_afrag_f32(const float* base, int m, int hh, int kb) {
    const float* ap = base + m * K + kb * 32 + hh * 8;
    float4 f0 = *(const float4*)ap;
    float4 f1 = *(const float4*)(ap + 4);
    float4 f2 = *(const float4*)(ap + 16);
    float4 f3 = *(const float4*)(ap + 20);
    v16h r;
    r[0]  = (_Float16)f0.x; r[1]  = (_Float16)f0.y; r[2]  = (_Float16)f0.z; r[3]  = (_Float16)f0.w;
    r[4]  = (_Float16)f1.x; r[5]  = (_Float16)f1.y; r[6]  = (_Float16)f1.z; r[7]  = (_Float16)f1.w;
    r[8]  = (_Float16)f2.x; r[9]  = (_Float16)f2.y; r[10] = (_Float16)f2.z; r[11] = (_Float16)f2.w;
    r[12] = (_Float16)f3.x; r[13] = (_Float16)f3.y; r[14] = (_Float16)f3.z; r[15] = (_Float16)f3.w;
    return r;
}

// ---------------- MLP layer (per-wave, 2 row-tiles) ----------------
// in_lds : [32][K] row-major, f16 (AF32=false) or f32 (AF32=true), wave-private
// out_lds: [32][G] f16 row-major, wave-private
// Wt     : [K][G] f16 (transposed weights, global/L2)
template<int K, int G, bool RELU, bool AF32>
__device__ void mlp_layer2(const void* __restrict__ in_lds,
                           _Float16* __restrict__ out_lds,
                           const _Float16* __restrict__ Wt,
                           const float* __restrict__ bias,
                           int lane) {
    const int m  = lane & 15;
    const int hh = lane >> 4;
    constexpr int KB = K / 32;
    constexpr int NT = G / 16;

    // Preload A fragments for both row-tiles.
    v16h a0[KB], a1[KB];
#pragma unroll
    for (int kb = 0; kb < KB; ++kb) {
        if constexpr (AF32) {
            const float* in32 = (const float*)in_lds;
            a0[kb] = load_afrag_f32<K>(in32,          m, hh, kb);
            a1[kb] = load_afrag_f32<K>(in32 + 16 * K, m, hh, kb);
        } else {
            const _Float16* in16 = (const _Float16*)in_lds;
            a0[kb] = load_afrag<K>(in16,          m, hh, kb);
            a1[kb] = load_afrag<K>(in16 + 16 * K, m, hh, kb);
        }
    }

#pragma unroll 1
    for (int nt = 0; nt < NT; ++nt) {
        v8f c0 = {}, c1 = {};
#pragma unroll
        for (int kb = 0; kb < KB; ++kb) {
            // B fragment: column N = m, halves = K rows kb*32 + hh*8 + [0..7] and kb*32 + 16 + hh*8 + [0..7].
            const _Float16* bp = Wt + (size_t)(kb * 32 + hh * 8) * G + nt * 16 + m;
            Pack16 pb;
#pragma unroll
            for (int s = 0; s < 8; ++s) { pb.v[s] = bp[s * G]; pb.v[8 + s] = bp[(16 + s) * G]; }
            // One B load feeds two WMMAs (register-level B reuse).
            c0 = __builtin_amdgcn_wmma_f32_16x16x32_f16(
                     false, a0[kb], false, pb.v, (short)0, c0, false, false);
            c1 = __builtin_amdgcn_wmma_f32_16x16x32_f16(
                     false, a1[kb], false, pb.v, (short)0, c1, false, false);
        }
        float bv = bias[nt * 16 + m];
#pragma unroll
        for (int j = 0; j < 8; ++j) {
            float v0 = c0[j] + bv;
            float v1 = c1[j] + bv;
            if (RELU) {
                v0 = (v0 > 0.0f) ? v0 : 0.0f;
                v1 = (v1 > 0.0f) ? v1 : 0.0f;
            }
            // C layout: row M = hh*8 + j, col N = m
            out_lds[(hh * 8 + j) * G + nt * 16 + m]      = (_Float16)v0;
            out_lds[(16 + hh * 8 + j) * G + nt * 16 + m] = (_Float16)v1;
        }
    }
}

// ---------------- main fused kernel ----------------

__global__ __launch_bounds__(WAVES_PER_BLOCK * 32)
void mlp_pool_kernel(const float* __restrict__ x,
                     const float* __restrict__ b1, const float* __restrict__ b2,
                     const float* __restrict__ b3, const float* __restrict__ b4,
                     const float* __restrict__ alpha,
                     const _Float16* __restrict__ Wt1, const _Float16* __restrict__ Wt2,
                     const _Float16* __restrict__ Wt3, const _Float16* __restrict__ Wt4,
                     float* __restrict__ accum, int ntiles) {
    // Per-wave ping-pong buffers. ldsB doubles as the f32 x-stage (32*128*4 = 16384 B
    // == 32*256*2 B) and as the f16 h2 buffer.
    __shared__ alignas(16) _Float16 ldsA[WAVES_PER_BLOCK][ROWS_PER_WAVE * 384];
    __shared__ alignas(16) _Float16 ldsB[WAVES_PER_BLOCK][ROWS_PER_WAVE * 256];

    const int lane = threadIdx.x & 31;
    const int wv   = threadIdx.x >> 5;
    const int tile = blockIdx.x * WAVES_PER_BLOCK + wv;
    if (tile >= ntiles) return;               // wave-uniform
    const int row0 = tile * ROWS_PER_WAVE;

    _Float16* bufA  = ldsA[wv];
    _Float16* bufB  = ldsB[wv];
    float*    bufX  = (float*)bufB;

    const int m  = lane & 15;
    const int hh = lane >> 4;

    // Stage x tile (32 rows x 128 f32) into LDS via the Tensor Data Mover.
    tdm_load_tile_f32(x + (size_t)row0 * FEAT, bufX, ROWS_PER_WAVE, FEAT);

    mlp_layer2<128, 384, false, true >(bufX, bufA, Wt1, b1, lane);   // h1 (f32 A input)
    mlp_layer2<384, 256, false, false>(bufA, bufB, Wt2, b2, lane);   // h2
    mlp_layer2<256, 128, true , false>(bufB, bufA, Wt3, b3, lane);   // h3 = relu(...)

    // Layer 4: p = h3 @ W4^T + b4  (G padded 5 -> 16), then pooling partials.
    {
        constexpr int K = 128, G = 16, KB = K / 32;
        v16h a0[KB], a1[KB];
#pragma unroll
        for (int kb = 0; kb < KB; ++kb) {
            a0[kb] = load_afrag<K>(bufA,          m, hh, kb);
            a1[kb] = load_afrag<K>(bufA + 16 * K, m, hh, kb);
        }
        v8f c0 = {}, c1 = {};
#pragma unroll
        for (int kb = 0; kb < KB; ++kb) {
            const _Float16* bp = Wt4 + (size_t)(kb * 32 + hh * 8) * G + m;
            Pack16 pb;
#pragma unroll
            for (int s = 0; s < 8; ++s) { pb.v[s] = bp[s * G]; pb.v[8 + s] = bp[(16 + s) * G]; }
            c0 = __builtin_amdgcn_wmma_f32_16x16x32_f16(
                     false, a0[kb], false, pb.v, (short)0, c0, false, false);
            c1 = __builtin_amdgcn_wmma_f32_16x16x32_f16(
                     false, a1[kb], false, pb.v, (short)0, c1, false, false);
        }

        float bv = (m < 5) ? b4[m]    : 0.0f;
        float al = (m < 5) ? alpha[m] : 0.0f;
        float s = 0.0f, t = 0.0f;
#pragma unroll
        for (int j = 0; j < 8; ++j) {
            float p0 = c0[j] + bv;                // p[row0 + hh*8 + j][m]
            float p1 = c1[j] + bv;                // p[row0 + 16 + hh*8 + j][m]
            float e0 = __expf(al * p0);           // softmax shift-invariant; |al*p| small
            float e1 = __expf(al * p1);
            s += e0 + e1;
            t += p0 * e0 + p1 * e1;
        }
        // combine the two lane-halves holding the same output column m
        s += __shfl_xor(s, 16, 32);
        t += __shfl_xor(t, 16, 32);
        if (hh == 0 && m < 5) {
            atomicAdd(&accum[m], s);       // S_o
            atomicAdd(&accum[8 + m], t);   // T_o
        }
    }
}

__global__ void finalize_kernel(const float* __restrict__ accum, float* __restrict__ out) {
    int i = threadIdx.x;
    if (i < 5) out[i] = accum[8 + i] / accum[i];
}

// ---------------- launch ----------------

extern "C" void kernel_launch(void* const* d_in, const int* in_sizes, int n_in,
                              void* d_out, int out_size, void* d_ws, size_t ws_size,
                              hipStream_t stream) {
    (void)n_in; (void)out_size; (void)ws_size;

    const float* x     = (const float*)d_in[0];
    const float* W1    = (const float*)d_in[1];
    const float* b1    = (const float*)d_in[2];
    const float* W2    = (const float*)d_in[3];
    const float* b2    = (const float*)d_in[4];
    const float* W3    = (const float*)d_in[5];
    const float* b3    = (const float*)d_in[6];
    const float* W4    = (const float*)d_in[7];
    const float* b4    = (const float*)d_in[8];
    const float* alpha = (const float*)d_in[9];
    float* out = (float*)d_out;

    char* ws = (char*)d_ws;
    float*    accum = (float*)(ws + WS_ACCUM);
    _Float16* Wt1   = (_Float16*)(ws + WS_WT1);
    _Float16* Wt2   = (_Float16*)(ws + WS_WT2);
    _Float16* Wt3   = (_Float16*)(ws + WS_WT3);
    _Float16* Wt4   = (_Float16*)(ws + WS_WT4);

    const int N      = in_sizes[0] / FEAT;        // rows
    const int ntiles = N / ROWS_PER_WAVE;         // 200000 / 32 = 6250

    init_accum_kernel<<<1, 32, 0, stream>>>(accum);

    {   // W1 [384][128] -> Wt1 [128][384]
        int tot = 128 * 384;
        convert_wt_kernel<<<(tot + 255) / 256, 256, 0, stream>>>(W1, Wt1, 384, 128, 384);
    }
    {   // W2 [256][384] -> Wt2 [384][256]
        int tot = 384 * 256;
        convert_wt_kernel<<<(tot + 255) / 256, 256, 0, stream>>>(W2, Wt2, 256, 384, 256);
    }
    {   // W3 [128][256] -> Wt3 [256][128]
        int tot = 256 * 128;
        convert_wt_kernel<<<(tot + 255) / 256, 256, 0, stream>>>(W3, Wt3, 128, 256, 128);
    }
    {   // W4 [5][128] -> Wt4 [128][16] zero-padded
        int tot = 128 * 16;
        convert_wt_kernel<<<(tot + 255) / 256, 256, 0, stream>>>(W4, Wt4, 5, 128, 16);
    }

    const int nblocks = (ntiles + WAVES_PER_BLOCK - 1) / WAVES_PER_BLOCK;
    mlp_pool_kernel<<<nblocks, WAVES_PER_BLOCK * 32, 0, stream>>>(
        x, b1, b2, b3, b4, alpha, Wt1, Wt2, Wt3, Wt4, accum, ntiles);

    finalize_kernel<<<1, 32, 0, stream>>>(accum, out);
}
